// GAT_85993835200537
// MI455X (gfx1250) — matrix operand order repaired and checked by reading, not compiled, not run on hardware
//
#include <hip/hip_runtime.h>
#include <math.h>

#define FEATS 128
#define HEADS 8
#define CDIM  16
#define NSLOPE 0.2f
#define EPSV  1e-5f
#define NLAYERS 5

typedef __attribute__((ext_vector_type(16))) _Float16 v16h;
typedef __attribute__((ext_vector_type(8)))  _Float16 v8h;
typedef __attribute__((ext_vector_type(8)))  float    v8f;
typedef __attribute__((ext_vector_type(4)))  int      v4i;

// ---- CDNA5 async global->LDS copy (ASYNCcnt path), with portable fallback ----
// Probe-confirmed: builtin exists; arg0 is a non-const AS(1) pointer to int4.
#if __has_builtin(__builtin_amdgcn_global_load_async_to_lds_b128)
#define ASYNC_COPY16(l, g)                                                     \
    __builtin_amdgcn_global_load_async_to_lds_b128(                            \
        (__attribute__((address_space(1))) v4i*)(void*)(g),                    \
        (__attribute__((address_space(3))) v4i*)(void*)(l), 0, 0)
#if __has_builtin(__builtin_amdgcn_s_wait_asynccnt)
#define WAIT_ASYNC() __builtin_amdgcn_s_wait_asynccnt(0)
#else
#define WAIT_ASYNC() asm volatile("s_wait_asynccnt 0" ::: "memory")
#endif
#else
#define ASYNC_COPY16(l, g) (*(v8h*)(l) = *(const v8h*)(g))
#define WAIT_ASYNC() ((void)0)
#endif

__device__ __forceinline__ float leakyf(float x) { return x > 0.f ? x : NSLOPE * x; }

// atomic max for float via monotonic int/uint trick (emax initialized to -inf)
__device__ __forceinline__ void atomicMaxF(float* a, float v) {
    if (v >= 0.f) atomicMax((int*)a, __float_as_int(v));
    else          atomicMin((unsigned int*)a, (unsigned int)__float_as_int(v));
}

// ---------------------------------------------------------------------------
// WMMA GEMM: H[n x NCOLS] = X[n x 128] * W[128 x NCOLS] (+ bias)
// X is a pre-converted f16 matrix [n][128]; W is pre-converted & pre-transposed
// f16 [NCOLS][128] (k contiguous). Staging into LDS is a pure byte copy done
// with GLOBAL_LOAD_ASYNC_TO_LDS_B128 when available.
// block = 128 threads (4 waves), each block computes 64 rows x NCOLS cols.
// ---------------------------------------------------------------------------
template<int NCOLS, bool HAS_BIAS>
__launch_bounds__(128)
__global__ void k_gemm_wmma(const _Float16* __restrict__ X16,
                            const _Float16* __restrict__ W16t,
                            const float* __restrict__ bias,
                            float* __restrict__ Hout, int n)
{
    constexpr int LDR = FEATS + 8; // 136 halves per row => 272B pitch (17 dwords*4)
    __shared__ _Float16 Xs[64 * LDR];
    __shared__ _Float16 Ws[NCOLS * LDR];

    const int tid  = threadIdx.x;
    const int row0 = blockIdx.x * 64;
    const bool full = (row0 + 64) <= n;

    // stage W^T tile: NCOLS rows x 128 halves, in 16B chunks (16 chunks/row)
    for (int ck = tid; ck < NCOLS * 16; ck += 128) {
        int r = ck >> 4, c = ck & 15;
        ASYNC_COPY16(&Ws[r * LDR + c * 8], W16t + (size_t)r * FEATS + c * 8);
    }
    // stage X slab: 64 rows x 128 halves
    if (full) {
        for (int ck = tid; ck < 64 * 16; ck += 128) {
            int r = ck >> 4, c = ck & 15;
            ASYNC_COPY16(&Xs[r * LDR + c * 8],
                         X16 + (size_t)(row0 + r) * FEATS + c * 8);
        }
    } else {
        for (int ck = tid; ck < 64 * 16; ck += 128) {
            int r = ck >> 4, c = ck & 15;
            int gr = row0 + r;
            if (gr < n) {
                ASYNC_COPY16(&Xs[r * LDR + c * 8],
                             X16 + (size_t)gr * FEATS + c * 8);
            } else {
                v8h z = {};
                *(v8h*)&Xs[r * LDR + c * 8] = z;
            }
        }
    }
    WAIT_ASYNC();
    __syncthreads();

    const int wv   = tid >> 5;   // wave 0..3 -> 16-row slab
    const int lane = tid & 31;
    const int mr   = lane & 15;
    const int hs   = lane >> 4;  // half-wave select
    const _Float16* xrow = &Xs[(wv * 16 + mr) * LDR];

    for (int nt = 0; nt < NCOLS / 16; ++nt) {
        const _Float16* brow = &Ws[(nt * 16 + mr) * LDR];
        v8f acc = {0.f, 0.f, 0.f, 0.f, 0.f, 0.f, 0.f, 0.f};
#pragma unroll
        for (int kt = 0; kt < FEATS / 32; ++kt) {
            // A fragment (ISA dense 16-bit A 16x32 layout):
            //  lanes 0-15: K 0..7 then 16..23 ; lanes 16-31: K 8..15 then 24..31
            const int ka = kt * 32 + hs * 8;
            v8h alo = *(const v8h*)(xrow + ka);
            v8h ahi = *(const v8h*)(xrow + ka + 16);
            // B fragment (same K order as A):
            //  lanes 0-15: K 0..7 then 16..23 ; lanes 16-31: K 8..15 then 24..31, n = lane&15
            const int kb = kt * 32 + hs * 8;
            v8h blo = *(const v8h*)(brow + kb);
            v8h bhi = *(const v8h*)(brow + kb + 16);
            v16h a = __builtin_shufflevector(alo, ahi, 0,1,2,3,4,5,6,7,8,9,10,11,12,13,14,15);
            v16h b = __builtin_shufflevector(blo, bhi, 0,1,2,3,4,5,6,7,8,9,10,11,12,13,14,15);
            acc = __builtin_amdgcn_wmma_f32_16x16x32_f16(false, a, false, b,
                                                         (short)0, acc, false, false);
        }
        // C/D layout: VGPR r -> M = r (lanes 0-15) / r+8 (lanes 16-31); N = lane&15
        const int gm0  = row0 + wv * 16 + hs * 8;
        const int ncol = nt * 16 + mr;
        float bv = 0.f;
        if (HAS_BIAS) bv = bias[ncol];
        if (full) {
#pragma unroll
            for (int r = 0; r < 8; ++r)
                Hout[(size_t)(gm0 + r) * NCOLS + ncol] = acc[r] + bv;
        } else {
#pragma unroll
            for (int r = 0; r < 8; ++r) {
                int gr = gm0 + r;
                if (gr < n) Hout[(size_t)gr * NCOLS + ncol] = acc[r] + bv;
            }
        }
    }
}

// -------------------- weight pre-convert / pre-transpose -------------------
// w16 layout: [5][ncol=128][k=128] f16 (transposed), then lin [ncol=64][k=128]
__global__ void k_prep_w(const float* __restrict__ gat_w,
                         const float* __restrict__ lin_w, _Float16* w16) {
    int idx = blockIdx.x * blockDim.x + threadIdx.x;
    const int totalGat = NLAYERS * FEATS * FEATS;
    if (idx < totalGat) {
        int l = idx / (FEATS * FEATS);
        int rem = idx - l * (FEATS * FEATS);
        int ncol = rem >> 7, k = rem & 127;
        w16[idx] = (_Float16)gat_w[(size_t)l * FEATS * FEATS + k * FEATS + ncol];
    } else if (idx < totalGat + 64 * FEATS) {
        int rem = idx - totalGat;
        int ncol = rem >> 7, k = rem & 127;
        w16[idx] = (_Float16)lin_w[k * 64 + ncol];
    }
}

// ------------------------------- GCN stage ---------------------------------
__global__ void k_deg_init(float* deg, int n) {
    int i = blockIdx.x * blockDim.x + threadIdx.x;
    if (i < n) deg[i] = 1.f;  // self-loop
}
__global__ void k_deg_edge(const int* __restrict__ ei, float* deg, int E) {
    int e = blockIdx.x * blockDim.x + threadIdx.x;
    if (e < E) atomicAdd(&deg[ei[E + e]], 1.f);
}
__global__ void k_gcn_self(const float* __restrict__ sig, const float* __restrict__ deg,
                           float* dinv, float* tsum, int n) {
    int i = blockIdx.x * blockDim.x + threadIdx.x;
    if (i < n) {
        float dv = rsqrtf(deg[i]);
        dinv[i] = dv;
        tsum[i] = dv * dv * sig[i];  // self-loop contribution
    }
}
__global__ void k_gcn_edge(const int* __restrict__ ei, const float* __restrict__ sig,
                           const float* __restrict__ dinv, float* tsum, int E) {
    int e = blockIdx.x * blockDim.x + threadIdx.x;
    if (e < E) {
        int s = ei[e], d = ei[E + e];
        atomicAdd(&tsum[d], dinv[s] * dinv[d] * sig[s]);
    }
}
__global__ void k_gcn_feat(const float* __restrict__ tsum, const float* __restrict__ w,
                           const float* __restrict__ b, float* x, _Float16* x16, int n) {
    int idx = blockIdx.x * blockDim.x + threadIdx.x;
    if (idx < n * FEATS) {
        int c = idx & 127;
        float v = fmaxf(tsum[idx >> 7] * w[c] + b[c], 0.f);
        x[idx] = v;
        x16[idx] = (_Float16)v;
    }
}

// ------------------------------ GraphNorm ----------------------------------
__global__ void k_zero(float* p, int cnt) {
    int i = blockIdx.x * blockDim.x + threadIdx.x;
    if (i < cnt) p[i] = 0.f;
}
__global__ void k_colsum(const float* __restrict__ x, float* cs, float* cq, int n) {
    int c = threadIdx.x;          // 128 threads: one column each
    float s = 0.f, sq = 0.f;
    for (int r = blockIdx.x; r < n; r += gridDim.x) {
        float v = x[(size_t)r * FEATS + c];
        s += v; sq += v * v;
    }
    atomicAdd(&cs[c], s);
    atomicAdd(&cq[c], sq);
}
__global__ void k_gnorm(float* x, _Float16* x16,
                        const float* __restrict__ cs, const float* __restrict__ cq,
                        const float* __restrict__ w, const float* __restrict__ b,
                        const float* __restrict__ sc, int n) {
    int idx = blockIdx.x * blockDim.x + threadIdx.x;
    if (idx < n * FEATS) {
        int c = idx & 127;
        float inv_n = 1.f / (float)n;
        float mu  = cs[c] * inv_n;
        float ex2 = cq[c] * inv_n;
        float s   = sc[c];
        // E[(x - s*mu)^2] = E[x^2] - (2s - s^2) mu^2
        float var = ex2 - (2.f * s - s * s) * mu * mu;
        float xc  = x[idx] - s * mu;
        float v = w[c] * xc * rsqrtf(var + EPSV) + b[c];
        x[idx] = v;
        x16[idx] = (_Float16)v;
    }
}

// ----------------------------- GAT edge phase ------------------------------
__global__ void k_att(const float* __restrict__ h, const float* __restrict__ ws,
                      const float* __restrict__ wd, float* a_s, float* a_d, int n) {
    int idx = blockIdx.x * blockDim.x + threadIdx.x;
    if (idx < n * HEADS) {
        int hd = idx & 7, node = idx >> 3;
        const float* hp = &h[(size_t)node * FEATS + hd * CDIM];
        const float* ps = &ws[hd * CDIM];
        const float* pd = &wd[hd * CDIM];
        float ss = 0.f, sd = 0.f;
#pragma unroll
        for (int c = 0; c < CDIM; ++c) { float v = hp[c]; ss += v * ps[c]; sd += v * pd[c]; }
        a_s[idx] = ss; a_d[idx] = sd;
    }
}
__global__ void k_att_init(float* emax, float* den, int n) {
    int i = blockIdx.x * blockDim.x + threadIdx.x;
    if (i < n * HEADS) { emax[i] = __int_as_float(0xff800000); den[i] = 0.f; }
}
__global__ void k_edge_max(const int* __restrict__ ei, const float* __restrict__ a_s,
                           const float* __restrict__ a_d, float* emax, int E, int n) {
    int e = blockIdx.x * blockDim.x + threadIdx.x;
    if (e >= E + n) return;
    int s, d;
    if (e < E) { s = ei[e]; d = ei[E + e]; } else { s = d = e - E; }
#pragma unroll
    for (int hd = 0; hd < HEADS; ++hd) {
        float v = leakyf(a_s[s * HEADS + hd] + a_d[d * HEADS + hd]);
        atomicMaxF(&emax[d * HEADS + hd], v);
    }
}
__global__ void k_edge_sum(const int* __restrict__ ei, const float* __restrict__ a_s,
                           const float* __restrict__ a_d, const float* __restrict__ emax,
                           float* den, int E, int n) {
    int e = blockIdx.x * blockDim.x + threadIdx.x;
    if (e >= E + n) return;
    int s, d;
    if (e < E) { s = ei[e]; d = ei[E + e]; } else { s = d = e - E; }
#pragma unroll
    for (int hd = 0; hd < HEADS; ++hd) {
        float v = leakyf(a_s[s * HEADS + hd] + a_d[d * HEADS + hd]);
        atomicAdd(&den[d * HEADS + hd], __expf(v - emax[d * HEADS + hd]));
    }
}
// one wave per edge: lane handles 4 consecutive features (float4), L2-resident gather
__global__ void k_edge_acc(const int* __restrict__ ei, const float* __restrict__ a_s,
                           const float* __restrict__ a_d, const float* __restrict__ emax,
                           const float* __restrict__ den, const float* __restrict__ h,
                           float* out, int E, int n) {
    long long tid = (long long)blockIdx.x * blockDim.x + threadIdx.x;
    int e  = (int)(tid >> 5);
    int ln = (int)(tid & 31);
    if (e >= E + n) return;
    int s, d;
    if (e < E) { s = ei[e]; d = ei[E + e]; } else { s = d = e - E; }
    int f0 = ln * 4;
    int hd = f0 >> 4;
    float v     = leakyf(a_s[s * HEADS + hd] + a_d[d * HEADS + hd]);
    float alpha = __expf(v - emax[d * HEADS + hd]) / den[d * HEADS + hd];
    const float4 hv = *(const float4*)&h[(size_t)s * FEATS + f0];
    float* op = &out[(size_t)d * FEATS + f0];
    atomicAdd(op + 0, alpha * hv.x);
    atomicAdd(op + 1, alpha * hv.y);
    atomicAdd(op + 2, alpha * hv.z);
    atomicAdd(op + 3, alpha * hv.w);
}
__global__ void k_update(float* x, _Float16* x16, const float* __restrict__ out,
                         const float* __restrict__ b, int n) {
    int idx = blockIdx.x * blockDim.x + threadIdx.x;
    if (idx < n * FEATS) {
        int c = idx & 127;
        float v = x[idx] + fmaxf(out[idx] + b[c], 0.f);
        x[idx] = v;
        x16[idx] = (_Float16)v;
    }
}

// ---------------------------------------------------------------------------
extern "C" void kernel_launch(void* const* d_in, const int* in_sizes, int n_in,
                              void* d_out, int out_size, void* d_ws, size_t ws_size,
                              hipStream_t stream)
{
    (void)n_in; (void)out_size; (void)ws_size;
    const float* signals = (const float*)d_in[0];
    const int*   ei      = (const int*)  d_in[1];
    const float* gcn_w   = (const float*)d_in[2];
    const float* gcn_b   = (const float*)d_in[3];
    const float* gat_w   = (const float*)d_in[4];
    const float* att_src = (const float*)d_in[5];
    const float* att_dst = (const float*)d_in[6];
    const float* gat_b   = (const float*)d_in[7];
    const float* gn_w    = (const float*)d_in[8];
    const float* gn_b    = (const float*)d_in[9];
    const float* gn_s    = (const float*)d_in[10];
    const float* lin_w   = (const float*)d_in[11];
    const float* lin_b   = (const float*)d_in[12];

    const int n = in_sizes[0];       // N_NODES
    const int E = in_sizes[1] / 2;   // N_EDGES

    // workspace carve-out (float units, 256B-aligned chunks)
    float* ws = (float*)d_ws;
    size_t off = 0;
    auto alloc = [&](size_t elems) {
        float* p = ws + off;
        off += (elems + 63) & ~(size_t)63;
        return p;
    };
    float* xbuf = alloc((size_t)n * FEATS);
    float* hbuf = alloc((size_t)n * FEATS);
    float* obuf = alloc((size_t)n * FEATS);
    float* a_s  = alloc((size_t)n * HEADS);
    float* a_d  = alloc((size_t)n * HEADS);
    float* emax = alloc((size_t)n * HEADS);
    float* den  = alloc((size_t)n * HEADS);
    float* deg  = alloc(n);
    float* dinv = alloc(n);
    float* tsum = alloc(n);
    float* csum = alloc(FEATS);   // csq contiguous right after
    float* csq  = alloc(FEATS);
    _Float16* x16 = (_Float16*)alloc((size_t)n * FEATS / 2);            // n*128 halves
    const int totalGatW = NLAYERS * FEATS * FEATS;                      // halves
    _Float16* w16 = (_Float16*)alloc((size_t)(totalGatW + 64 * FEATS) / 2 + 64);

    const int TB = 256;
    auto blks = [&](long long t) { return (int)((t + TB - 1) / TB); };
    const int gemmBlocks = (n + 63) / 64;

    // ---- pre-convert & transpose all GEMM weights to f16 (once per launch) ----
    k_prep_w<<<blks(totalGatW + 64 * FEATS), TB, 0, stream>>>(gat_w, lin_w, w16);

    // ---- GCNConv(1 -> 128), symmetric norm with self-loops ----
    k_deg_init<<<blks(n), TB, 0, stream>>>(deg, n);
    k_deg_edge<<<blks(E), TB, 0, stream>>>(ei, deg, E);
    k_gcn_self<<<blks(n), TB, 0, stream>>>(signals, deg, dinv, tsum, n);
    k_gcn_edge<<<blks(E), TB, 0, stream>>>(ei, signals, dinv, tsum, E);
    k_gcn_feat<<<blks((long long)n * FEATS), TB, 0, stream>>>(tsum, gcn_w, gcn_b, xbuf, x16, n);

    // ---- 5 GAT layers ----
    for (int l = 0; l < NLAYERS; ++l) {
        if (l % 3 == 0) {
            int g = l / 3;
            k_zero<<<1, 256, 0, stream>>>(csum, 2 * FEATS);   // csum + csq
            k_colsum<<<512, 128, 0, stream>>>(xbuf, csum, csq, n);
            k_gnorm<<<blks((long long)n * FEATS), TB, 0, stream>>>(
                xbuf, x16, csum, csq, gn_w + g * FEATS, gn_b + g * FEATS, gn_s + g * FEATS, n);
        }
        // h = x @ gat_w[l]   (WMMA f16->f32, async LDS staging)
        k_gemm_wmma<FEATS, false><<<gemmBlocks, 128, 0, stream>>>(
            x16, w16 + (size_t)l * FEATS * FEATS, nullptr, hbuf, n);
        // per-node attention logits
        k_att<<<blks((long long)n * HEADS), TB, 0, stream>>>(
            hbuf, att_src + l * HEADS * CDIM, att_dst + l * HEADS * CDIM, a_s, a_d, n);
        // init segment-softmax state + output accumulator
        k_att_init<<<blks((long long)n * HEADS), TB, 0, stream>>>(emax, den, n);
        k_zero<<<blks((long long)n * FEATS), TB, 0, stream>>>(obuf, n * FEATS);
        // edge softmax (3 passes, self-loops appended as edges E..E+n-1)
        k_edge_max<<<blks((long long)(E + n)), TB, 0, stream>>>(ei, a_s, a_d, emax, E, n);
        k_edge_sum<<<blks((long long)(E + n)), TB, 0, stream>>>(ei, a_s, a_d, emax, den, E, n);
        k_edge_acc<<<blks((long long)(E + n) * 32), TB, 0, stream>>>(
            ei, a_s, a_d, emax, den, hbuf, obuf, E, n);
        // residual update (also refresh f16 copy of x)
        k_update<<<blks((long long)n * FEATS), TB, 0, stream>>>(
            xbuf, x16, obuf, gat_b + l * FEATS, n);
    }

    // ---- final linear 128 -> 64 (WMMA) straight into d_out ----
    k_gemm_wmma<64, true><<<gemmBlocks, 128, 0, stream>>>(
        x16, w16 + totalGatW, lin_b, (float*)d_out, n);
}
